// CustomModel_71820443123928
// MI455X (gfx1250) — matrix operand order repaired and checked by reading, not compiled, not run on hardware
//
#include <hip/hip_runtime.h>
#include <hip/hip_bf16.h>
#include <stdint.h>
#include <stddef.h>

#define VOCAB 30522
#define HDIM  768
#define NLAY  12
#define TSEQ  128
#define BATCH 32
#define G3H   (3 * HDIM)   // 2304

typedef __attribute__((ext_vector_type(16))) __bf16 v16bf;
typedef __attribute__((ext_vector_type(8)))  __bf16 v8bf;
typedef __attribute__((ext_vector_type(8)))  float  v8f;
typedef __attribute__((ext_vector_type(4)))  unsigned int v4u;
typedef __attribute__((ext_vector_type(8)))  int v8i;
typedef __attribute__((ext_vector_type(4)))  int v4i;

// ---------------------------------------------------------------------------
// WMMA fragment builders (wave32; contiguous-K order used here, not the ISA table order).
// A (16x32 MxK): lane r<16 -> row r, K{0..15}; lane r+16 -> K{16..31}
// B (32x16 KxN): lane r<16 -> col r, K{0..15};       lane r+16 -> K{16..31}
// Both are contiguous-K slices when the source is K-major.
// ---------------------------------------------------------------------------
__device__ __forceinline__ v16bf frag_pair(const __bf16* p0, const __bf16* p1) {
  v8bf a = *(const v8bf*)p0;
  v8bf b = *(const v8bf*)p1;
  return __builtin_shufflevector(a, b, 0, 1, 2, 3, 4, 5, 6, 7,
                                       8, 9, 10, 11, 12, 13, 14, 15);
}

// ---------------------------------------------------------------------------
// Tensor Data Mover: 2-D tile (tile_rows x 32 bf16) global -> LDS, with a
// 16-byte pad after each 64-byte row (bank-conflict avoidance) and hardware
// zero-fill for rows/cols beyond tensor_dim (handles the decoder N-tail).
// D# layout per CDNA5 ISA. This toolchain uses
// the 6-arg builtin form: (g0, g1, g2, g3, extra_group, cpol).
// ---------------------------------------------------------------------------
#define TDM_TILE_COLS 32          // K elements per tile row
#define LDS_STRIDE    40          // 32 elems + 8 elems (16B) pad, in bf16

__device__ __forceinline__ unsigned lds_offset(const void* p) {
  // LDS aperture keeps the byte offset in the low 32 bits of a generic pointer
  return (unsigned)(uintptr_t)p;
}

__device__ __forceinline__ void tdm_load_tile(const __bf16* gaddr,
                                              unsigned lds_addr,
                                              unsigned tensor_rows,
                                              unsigned tensor_cols,  // = K
                                              unsigned tile_rows) {
#if __has_builtin(__builtin_amdgcn_tensor_load_to_lds)
  union { unsigned u[4]; v4u v; } g0;
  union { int i[8]; v8i v; } g1;
  const uint64_t ga = (uint64_t)(uintptr_t)gaddr;

  g0.u[0] = 1u;                                   // count=1, is_restore=0
  g0.u[1] = lds_addr;                             // lds_addr (bytes)
  g0.u[2] = (unsigned)ga;                         // global_addr[31:0]
  g0.u[3] = (unsigned)(ga >> 32) | (2u << 30);    // global_addr[56:32], type=2

  const unsigned td0 = tensor_cols;               // tensor_dim0 (elems)
  const unsigned td1 = tensor_rows;               // tensor_dim1 (rows)
  const uint64_t st0 = (uint64_t)tensor_cols;     // dim0 stride (elems)

  // word0: data_size=1 (2B) @16, pad_enable @20, pad_interval=3 (64B) @24:22,
  //        pad_amount=3 (4 DWORDs = 16B) @31:25
  g1.i[0] = (int)((1u << 16) | (1u << 20) | (3u << 22) | (3u << 25));
  g1.i[1] = (int)((td0 & 0xFFFFu) << 16);
  g1.i[2] = (int)((td0 >> 16) | ((td1 & 0xFFFFu) << 16));
  g1.i[3] = (int)((td1 >> 16) | ((unsigned)TDM_TILE_COLS << 16));
  g1.i[4] = (int)(tile_rows & 0xFFFFu);           // tile_dim1; tile_dim2=0
  g1.i[5] = (int)(st0 & 0xFFFFFFFFu);
  g1.i[6] = (int)((st0 >> 32) & 0xFFFFu);         // dim1 stride unused (2D)
  g1.i[7] = 0;

  v4i z4 = {0, 0, 0, 0};
  v8i z8 = {0, 0, 0, 0, 0, 0, 0, 0};
  __builtin_amdgcn_tensor_load_to_lds(g0.v, g1.v, z4, z4, z8, 0);
#else
  (void)gaddr; (void)lds_addr; (void)tensor_rows; (void)tensor_cols; (void)tile_rows;
#endif
}

// ---------------------------------------------------------------------------
// GEMM: C[M,N] (f32) = A[M,K] (bf16 row-major) * W[N,K]^T (bf16 row-major)
//       + bias[N].
// Block = 256 threads = 8 waves -> one 128x128 C tile.
// Per wave: 64(M) x 32(N) -> 4x2 accumulators, 8 WMMAs per K-step.
// A/B slabs (128 x 32 bf16, padded rows) staged in LDS by TDM, double-buffered.
// Each wave DMAs its own 16-row slice of each slab and waits TENSORcnt.
// ---------------------------------------------------------------------------
__global__ void __launch_bounds__(256)
gemm_bf16_wmma(const __bf16* __restrict__ A,
               const __bf16* __restrict__ W,
               const float*  __restrict__ bias,
               float*        __restrict__ C,
               int M, int N, int K) {
  __shared__ __bf16 ldsA[2][128 * LDS_STRIDE];
  __shared__ __bf16 ldsB[2][128 * LDS_STRIDE];

  const int lane = threadIdx.x & 31;
  const int wave = threadIdx.x >> 5;
  const int r    = lane & 15;
  const int hi   = lane >> 4;
  const int wm   = wave & 1;    // M half (0..1)  -> 64 rows
  const int wn   = wave >> 1;   // N quarter (0..3) -> 32 cols
  const int tm0  = blockIdx.y * 128;
  const int tn0  = blockIdx.x * 128;

  // per-wave 16-row DMA slices of the A and B slabs
  const int arow = tm0 + wave * 16;
  const int brow = tn0 + wave * 16;
  const unsigned sliceOff = (unsigned)(wave * 16 * LDS_STRIDE * sizeof(__bf16));

  v8f acc[4][2];
#pragma unroll
  for (int mt = 0; mt < 4; ++mt)
#pragma unroll
    for (int nt = 0; nt < 2; ++nt) acc[mt][nt] = (v8f){};

  // ---- prefill buffer 0 ----
  tdm_load_tile(A + (size_t)arow * K, lds_offset(ldsA[0]) + sliceOff,
                (unsigned)(M - arow), (unsigned)K, 16u);
  if (brow < N) {
    tdm_load_tile(W + (size_t)brow * K, lds_offset(ldsB[0]) + sliceOff,
                  (unsigned)(N - brow), (unsigned)K, 16u);
  }
  __builtin_amdgcn_s_wait_tensorcnt(0);
  __syncthreads();

  int buf = 0;
  for (int k0 = 0; k0 < K; k0 += TDM_TILE_COLS) {
    const int nbuf = buf ^ 1;
    // issue next-buffer DMAs; they overlap with this iteration's WMMAs
    if (k0 + TDM_TILE_COLS < K) {
      tdm_load_tile(A + (size_t)arow * K + (k0 + TDM_TILE_COLS),
                    lds_offset(ldsA[nbuf]) + sliceOff,
                    (unsigned)(M - arow), (unsigned)K, 16u);
      if (brow < N) {
        tdm_load_tile(W + (size_t)brow * K + (k0 + TDM_TILE_COLS),
                      lds_offset(ldsB[nbuf]) + sliceOff,
                      (unsigned)(N - brow), (unsigned)K, 16u);
      }
    }

    const __bf16* sA = ldsA[buf];
    const __bf16* sB = ldsB[buf];

    v16bf afrag[4];
#pragma unroll
    for (int mt = 0; mt < 4; ++mt) {
      const __bf16* p = sA + (size_t)(wm * 64 + mt * 16 + r) * LDS_STRIDE + hi * 16;
      afrag[mt] = frag_pair(p, p + 8);
    }
#pragma unroll
    for (int nt = 0; nt < 2; ++nt) {
      const __bf16* p = sB + (size_t)(wn * 32 + nt * 16 + r) * LDS_STRIDE + hi * 16;
      v16bf bfrag = frag_pair(p, p + 8);
#pragma unroll
      for (int mt = 0; mt < 4; ++mt) {
        acc[mt][nt] = __builtin_amdgcn_wmma_f32_16x16x32_bf16(
            false, afrag[mt], false, bfrag, (short)0, acc[mt][nt], false, false);
      }
    }

    __builtin_amdgcn_s_wait_tensorcnt(0);
    __syncthreads();
    buf = nbuf;
  }

  // C/D layout: VGPR i -> row = mtile + hi*8 + i, col = ntile + r
#pragma unroll
  for (int mt = 0; mt < 4; ++mt) {
#pragma unroll
    for (int nt = 0; nt < 2; ++nt) {
      const int col = tn0 + wn * 32 + nt * 16 + r;
      if (col >= N) continue;
      const float bv = bias ? bias[col] : 0.0f;
#pragma unroll
      for (int i = 0; i < 8; ++i) {
        const int row = tm0 + wm * 64 + mt * 16 + hi * 8 + i;
        C[(size_t)row * N + col] = acc[mt][nt][i] + bv;
      }
    }
  }
}

// ---------------------------------------------------------------------------
// Persistent GRU scan: SCAN_WGS workgroups, global generation-counter barrier.
// h kept in L2-resident workspace (f32 state + bf16 copy for WMMA).
// h_t written bf16 over the x buffer (gi was fully precomputed).
// ---------------------------------------------------------------------------
#define SCAN_WGS 18

__device__ __forceinline__ void grid_barrier(unsigned int* cnt,
                                             unsigned int nwg,
                                             unsigned int& phase) {
  __syncthreads();
  if (threadIdx.x == 0) {
    __threadfence();
    __hip_atomic_fetch_add(cnt, 1u, __ATOMIC_ACQ_REL, __HIP_MEMORY_SCOPE_AGENT);
    unsigned int target = phase + nwg;
    while (__hip_atomic_load(cnt, __ATOMIC_ACQUIRE,
                             __HIP_MEMORY_SCOPE_AGENT) < target) {
      __builtin_amdgcn_s_sleep(1);
    }
    phase = target;
    __threadfence();
  }
  __syncthreads();
}

__global__ void __launch_bounds__(256)
gru_scan(const __bf16*   __restrict__ whh,   // [3H, H] bf16, this layer
         const float*    __restrict__ b_hh,  // [3H]
         const float*    __restrict__ gi,    // [T, B, 3H] f32
         __hip_bfloat16* __restrict__ xout,  // [T, B, H] bf16 (overwritten)
         float*          __restrict__ gh,    // [B, 3H] scratch
         float*          __restrict__ hf,    // [B*H] f32 state
         __hip_bfloat16* __restrict__ hb,    // [B*H] bf16 state
         unsigned int*   __restrict__ cnt) {
  const unsigned int nwg = gridDim.x;
  const int nth  = (int)(blockDim.x * gridDim.x);
  const int gtid = (int)(blockIdx.x * blockDim.x + threadIdx.x);
  unsigned int phase = 0;

  for (int i = gtid; i < BATCH * HDIM; i += nth) {   // h0 = 0
    hf[i] = 0.0f;
    hb[i] = __float2bfloat16(0.0f);
  }
  grid_barrier(cnt, nwg, phase);

  const int lane   = threadIdx.x & 31;
  const int wave   = threadIdx.x >> 5;
  const int gwave  = (int)blockIdx.x * 8 + wave;
  const int nwaves = (int)gridDim.x * 8;
  const int r  = lane & 15;
  const int hi = lane >> 4;
  const __bf16* hbb = (const __bf16*)hb;

  for (int t = 0; t < TSEQ; ++t) {
    // gh = h * whh^T  (M=32, N=2304, K=768) -> 2 x 144 16x16 tiles
    for (int tile = gwave; tile < 2 * (G3H / 16); tile += nwaves) {
      const int mt = (tile & 1) * 16;
      const int nt = (tile >> 1) * 16;
      v8f acc = {};
      const __bf16* pa = hbb + (size_t)(mt + r) * HDIM + hi * 16;
      const __bf16* pb = whh + (size_t)(nt + r) * HDIM + hi * 16;
      for (int k0 = 0; k0 < HDIM; k0 += 32) {
        __builtin_prefetch(pb + 64, 0, 0);
        v16bf af = frag_pair(pa, pa + 8);
        v16bf bf = frag_pair(pb, pb + 8);
        acc = __builtin_amdgcn_wmma_f32_16x16x32_bf16(false, af, false, bf,
                                                      (short)0, acc,
                                                      false, false);
        pa += 32; pb += 32;
      }
#pragma unroll
      for (int i = 0; i < 8; ++i) {
        gh[(size_t)(mt + hi * 8 + i) * G3H + nt + r] = acc[i];
      }
    }
    grid_barrier(cnt, nwg, phase);

    // gate combine: r,z,n -> h_new
    const float* git = gi + (size_t)t * BATCH * G3H;
    for (int idx = gtid; idx < BATCH * HDIM; idx += nth) {
      const int b = idx / HDIM;
      const int j = idx - b * HDIM;
      const float* gib = git + (size_t)b * G3H;
      const float* ghb = gh + (size_t)b * G3H;
      float ir = gib[j], iz = gib[HDIM + j], inn = gib[2 * HDIM + j];
      float hr = ghb[j] + b_hh[j];
      float hz = ghb[HDIM + j] + b_hh[HDIM + j];
      float hn = ghb[2 * HDIM + j] + b_hh[2 * HDIM + j];
      float rg = 1.0f / (1.0f + __expf(-(ir + hr)));
      float zg = 1.0f / (1.0f + __expf(-(iz + hz)));
      float ng = tanhf(inn + rg * hn);
      float hnew = (1.0f - zg) * ng + zg * hf[idx];
      hf[idx] = hnew;
      __hip_bfloat16 hv = __float2bfloat16(hnew);
      hb[idx] = hv;
      xout[(size_t)t * BATCH * HDIM + idx] = hv;
    }
    grid_barrier(cnt, nwg, phase);
  }
}

// ---------------------------------------------------------------------------
// Helpers: f32 -> bf16 conversion, embedding gather (f32 table -> bf16 x).
// ---------------------------------------------------------------------------
__global__ void f32_to_bf16(const float* __restrict__ in,
                            __hip_bfloat16* __restrict__ out, int n) {
  int i = (int)(blockIdx.x * blockDim.x + threadIdx.x);
  int stride = (int)(gridDim.x * blockDim.x);
  for (; i < n; i += stride) out[i] = __float2bfloat16(in[i]);
}

__global__ void embed_gather(const int* __restrict__ ids,       // [T*B]
                             const float* __restrict__ emb,     // [VOCAB, H]
                             __hip_bfloat16* __restrict__ x) {  // [T*B, H]
  int i = (int)(blockIdx.x * blockDim.x + threadIdx.x);
  const int total = TSEQ * BATCH * HDIM;
  if (i >= total) return;
  int tb = i / HDIM;
  int j  = i - tb * HDIM;
  x[i] = __float2bfloat16(emb[(size_t)ids[tb] * HDIM + j]);
}

// ---------------------------------------------------------------------------
// Host-side launch
// ---------------------------------------------------------------------------
static inline size_t align256(size_t x) { return (x + 255u) & ~(size_t)255u; }

extern "C" void kernel_launch(void* const* d_in, const int* in_sizes, int n_in,
                              void* d_out, int out_size, void* d_ws, size_t ws_size,
                              hipStream_t stream) {
  const int*   ids   = (const int*)d_in[0];
  const float* emb   = (const float*)d_in[1];
  const float* w_ih  = (const float*)d_in[2];   // [L, 3H, H]
  const float* w_hh  = (const float*)d_in[3];   // [L, 3H, H]
  const float* b_ih  = (const float*)d_in[4];   // [L, 3H]
  const float* b_hh  = (const float*)d_in[5];   // [L, 3H]
  const float* dec_w = (const float*)d_in[6];   // [VOCAB, H]
  const float* dec_b = (const float*)d_in[7];   // [VOCAB]
  float* out = (float*)d_out;                   // [T, B, VOCAB]

  char* p = (char*)d_ws;
  auto take = [&](size_t bytes) { char* q = p; p += align256(bytes); return q; };

  const size_t wlayer = (size_t)G3H * HDIM;
  __hip_bfloat16* wih_bf = (__hip_bfloat16*)take(NLAY * wlayer * sizeof(__hip_bfloat16));
  __hip_bfloat16* whh_bf = (__hip_bfloat16*)take(NLAY * wlayer * sizeof(__hip_bfloat16));
  __hip_bfloat16* dec_bf = (__hip_bfloat16*)take((size_t)VOCAB * HDIM * sizeof(__hip_bfloat16));
  __hip_bfloat16* x_bf   = (__hip_bfloat16*)take((size_t)TSEQ * BATCH * HDIM * sizeof(__hip_bfloat16));
  float*          gi     = (float*)take((size_t)TSEQ * BATCH * G3H * sizeof(float));
  float*          gh     = (float*)take((size_t)BATCH * G3H * sizeof(float));
  float*          hf     = (float*)take((size_t)BATCH * HDIM * sizeof(float));
  __hip_bfloat16* hb     = (__hip_bfloat16*)take((size_t)BATCH * HDIM * sizeof(__hip_bfloat16));
  unsigned int*   cnts   = (unsigned int*)take(NLAY * sizeof(unsigned int));
  (void)ws_size; (void)in_sizes; (void)n_in; (void)out_size;

  (void)hipMemsetAsync(cnts, 0, NLAY * sizeof(unsigned int), stream);

  {
    int n = (int)(NLAY * wlayer);
    f32_to_bf16<<<dim3((n + 255) / 256), dim3(256), 0, stream>>>(w_ih, wih_bf, n);
    f32_to_bf16<<<dim3((n + 255) / 256), dim3(256), 0, stream>>>(w_hh, whh_bf, n);
    int nd = VOCAB * HDIM;
    f32_to_bf16<<<dim3((nd + 255) / 256), dim3(256), 0, stream>>>(dec_w, dec_bf, nd);
  }
  {
    int total = TSEQ * BATCH * HDIM;
    embed_gather<<<dim3((total + 255) / 256), dim3(256), 0, stream>>>(ids, emb, x_bf);
  }

  const int M = TSEQ * BATCH;  // 4096

  for (int l = 0; l < NLAY; ++l) {
    // gi = x * w_ih^T + b_ih    [4096, 2304]
    dim3 gridGi(G3H / 128, M / 128);
    gemm_bf16_wmma<<<gridGi, dim3(256), 0, stream>>>(
        (const __bf16*)x_bf,
        (const __bf16*)(wih_bf + (size_t)l * wlayer),
        b_ih + (size_t)l * G3H,
        gi, M, G3H, HDIM);

    gru_scan<<<dim3(SCAN_WGS), dim3(256), 0, stream>>>(
        (const __bf16*)(whh_bf + (size_t)l * wlayer),
        b_hh + (size_t)l * G3H,
        gi, x_bf, gh, hf, hb, cnts + l);
  }

  // logits = x * dec_w^T + dec_b   [4096, 30522]
  dim3 gridDec((VOCAB + 127) / 128, M / 128);
  gemm_bf16_wmma<<<gridDec, dim3(256), 0, stream>>>(
      (const __bf16*)x_bf, (const __bf16*)dec_bf, dec_b, out, M, VOCAB, HDIM);
}
